// TextEncoder_78228534329794
// MI455X (gfx1250) — matrix operand order repaired and checked by reading, not compiled, not run on hardware
//
#include <hip/hip_runtime.h>
#include <hip/hip_bf16.h>
#include <math.h>

#define BATCH  8
#define SEQ    256
#define DMODEL 1024
#define NHEAD  16
#define DHEAD  64
#define NLAYER 12
#define EMB    512
#define DFF    4096
#define MTOK   (BATCH * SEQ)   /* 2048 */

#define BM 128          /* block tile M */
#define BN 128          /* block tile N */
#define BK 32           /* k-chunk (one wmma K step) */

typedef __attribute__((ext_vector_type(16))) _Float16 v16h;
typedef __attribute__((ext_vector_type(8)))  _Float16 v8h;
typedef __attribute__((ext_vector_type(8)))  float    v8f;
typedef unsigned int u32x4 __attribute__((ext_vector_type(4)));
typedef int          i32x8 __attribute__((ext_vector_type(8)));
typedef int          i32x4 __attribute__((ext_vector_type(4)));

#if defined(__has_builtin)
#if __has_builtin(__builtin_amdgcn_tensor_load_to_lds) && __has_builtin(__builtin_amdgcn_s_wait_tensorcnt)
#define USE_TDM 1
#endif
#endif

// ---------------------------------------------------------------------------
// Fragment helpers; k order here is contiguous, not the ISA table order (CDNA5 ISA §7.12.2)
// A 16x32 f16: lane L (row M = L%16, grp = L/16) holds contiguous
//              K = grp*16..+15 (same k order as B).
// B 32x16 f16: lane L (col N = L%16) holds contiguous K = grp*16..+15.
// C 16x16 f32: VGPR r -> M = grp*8 + r, N = L%16.
// ---------------------------------------------------------------------------
static __device__ __forceinline__ v16h load_a_frag(const _Float16* row, int k0, int grp) {
  return *(const v16h*)(row + k0 + grp * 16);
}
static __device__ __forceinline__ v16h load_b_frag(const _Float16* row, int k0, int grp) {
  return *(const v16h*)(row + k0 + grp * 16);
}
static __device__ __forceinline__ v8f wmma_f16(v16h a, v16h b, v8f c) {
  return __builtin_amdgcn_wmma_f32_16x16x32_f16(false, a, false, b, (short)0, c, false, false);
}

#ifdef USE_TDM
// ---------------------------------------------------------------------------
// Tensor Data Mover: DMA a [rows x 64B] f16 tile (row stride = strideHalves)
// from global memory into LDS at ldsOff.  D# packing per CDNA5 ISA:
//  group0: count=1 | lds_addr | global_addr(57b) | type=2
//  group1: data_size=3 (8B) | tensor_dim0/1 (large) | tile_dim0=8 (8B units)
//          tile_dim1=rows | tensor_dim0_stride (48b, 8B units)
// 6-arg builtin form (clang-23 / therock-10.0 lane): groups 2,3 as int32x4,
// trailing int32x8 group, then cpol.
// ---------------------------------------------------------------------------
static __device__ __forceinline__ void tdm_load_tile(
    const _Float16* gsrc, unsigned ldsOff, int strideHalves, int rows)
{
  unsigned long long ga = (unsigned long long)(uintptr_t)gsrc;
  u32x4 g0;
  g0.x = 1u;                                        // count=1 (valid descriptor)
  g0.y = ldsOff;                                    // lds_addr (bytes)
  g0.z = (unsigned)(ga & 0xffffffffu);              // global_addr[31:0]
  g0.w = (unsigned)((ga >> 32) & 0x01ffffffu) | (2u << 30);  // addr[56:32] | type=2
  const unsigned td0 = 0x40000000u;                 // huge tensor dims (tile in-bounds)
  const unsigned td1 = 0x40000000u;
  unsigned long long s0 = (unsigned long long)(strideHalves >> 2);  // 8B units
  i32x8 g1;
  g1[0] = (int)(3u << 16);                                        // data_size=3 (8B)
  g1[1] = (int)((td0 & 0xffffu) << 16);                           // tensor_dim0 lo
  g1[2] = (int)(((td0 >> 16) & 0xffffu) | ((td1 & 0xffffu) << 16));
  g1[3] = (int)(((td1 >> 16) & 0xffffu) | (8u << 16));            // tile_dim0 = 8
  g1[4] = (int)(rows & 0xffffu);                                  // tile_dim1, tile_dim2=0
  g1[5] = (int)(s0 & 0xffffffffu);                                // dim0 stride lo
  g1[6] = (int)((s0 >> 32) & 0xffffu);                            // stride hi | dim1_stride lo = 0
  g1[7] = 0;
  i32x4 z4 = {0, 0, 0, 0};
  i32x8 z8 = {0, 0, 0, 0, 0, 0, 0, 0};
  __builtin_amdgcn_tensor_load_to_lds(g0, g1, z4, z4, z8, 0);
}
#endif

static __device__ __forceinline__ unsigned lds_off_u32(const void* p) {
  return (unsigned)(unsigned long long)(uintptr_t)p;   // low 32 bits = LDS offset
}

// ---------------------------------------------------------------------------
// Embedding: h[b,s,:] = w_voc[text[b,s]] + w_pos[s]
// ---------------------------------------------------------------------------
__global__ __launch_bounds__(256) void embed_kernel(
    const int* __restrict__ text, const float* __restrict__ wvoc,
    const float* __restrict__ wpos, float* __restrict__ h)
{
  size_t idx = (size_t)blockIdx.x * 256 + threadIdx.x;
  if (idx >= (size_t)MTOK * DMODEL) return;
  int d  = (int)(idx & (DMODEL - 1));
  int bs = (int)(idx >> 10);
  int s  = bs & (SEQ - 1);
  int tok = text[bs];
  h[idx] = wvoc[(size_t)tok * DMODEL + d] + wpos[(size_t)s * DMODEL + d];
}

// ---------------------------------------------------------------------------
// LayerNorm over D=1024, output f16 row-major [row, D]
// ---------------------------------------------------------------------------
__global__ __launch_bounds__(256) void layernorm_f16_kernel(
    const float* __restrict__ x, const float* __restrict__ g,
    const float* __restrict__ bta, _Float16* __restrict__ out)
{
  __shared__ float red[256];
  int row = blockIdx.x, tid = threadIdx.x;
  const float* xr = x + (size_t)row * DMODEL;
  float s = 0.f;
  for (int d = tid; d < DMODEL; d += 256) s += xr[d];
  red[tid] = s; __syncthreads();
  for (int off = 128; off > 0; off >>= 1) {
    if (tid < off) red[tid] += red[tid + off];
    __syncthreads();
  }
  float mu = red[0] * (1.0f / DMODEL);
  __syncthreads();
  float v = 0.f;
  for (int d = tid; d < DMODEL; d += 256) { float t = xr[d] - mu; v += t * t; }
  red[tid] = v; __syncthreads();
  for (int off = 128; off > 0; off >>= 1) {
    if (tid < off) red[tid] += red[tid + off];
    __syncthreads();
  }
  float rs = rsqrtf(red[0] * (1.0f / DMODEL) + 1e-5f);
  _Float16* orow = out + (size_t)row * DMODEL;
  for (int d = tid; d < DMODEL; d += 256)
    orow[d] = (_Float16)((xr[d] - mu) * rs * g[d] + bta[d]);
}

// ---------------------------------------------------------------------------
// fp32 weight [K,N] -> f16 transposed [N,K]
// ---------------------------------------------------------------------------
__global__ __launch_bounds__(256) void convert_wT_kernel(
    const float* __restrict__ W, _Float16* __restrict__ WT, int K, int N)
{
  size_t idx = (size_t)blockIdx.x * 256 + threadIdx.x;
  if (idx >= (size_t)K * N) return;
  size_t kk = idx / (size_t)N;
  size_t n  = idx - kk * (size_t)N;
  WT[n * (size_t)K + kk] = (_Float16)W[idx];
}

// ---------------------------------------------------------------------------
// WMMA GEMM: C[M=2048, N] = A[M,K]f16 @ B[K,N]  (B given as BT[N,K]f16)
// Block = 256 threads, 128x128 C tile; wave (2x4 grid) owns 64x32 region
// (8 accumulators, 8 WMMA per 32-K chunk).  A/BT tiles are staged into LDS
// by the Tensor Data Mover (wave 0 issues descriptors, TENSORcnt-pipelined,
// double buffered), all waves read fragments from LDS.
// Epilogues:
//  0: q -> [b,h,s,dh]+bias  1: k -> [b,h,s,dh]  2: vT -> [b,h,dh,s]+bias
//  3: h += val+bias (fp32)  4: gelu(val+bias) -> f16 [m,n]
// ---------------------------------------------------------------------------
__global__ __launch_bounds__(256) void gemm_f16_wmma_kernel(
    const _Float16* __restrict__ A, const _Float16* __restrict__ BT,
    const float* __restrict__ bias, float* __restrict__ outF,
    _Float16* __restrict__ out16, int N, int K, int epi)
{
  __shared__ _Float16 ldsA[2][BM][BK];   // 16 KB
  __shared__ _Float16 ldsB[2][BN][BK];   // 16 KB
  int tid  = threadIdx.x;
  int lane = tid & 31;
  int wid  = tid >> 5;
  int nbn  = N >> 7;
  int bm   = blockIdx.x / nbn;
  int bn   = blockIdx.x - bm * nbn;
  int m0   = bm << 7, n0 = bn << 7;
  int wm   = wid >> 2;          // 0..1 -> 64 rows each
  int wn   = wid & 3;           // 0..3 -> 32 cols each
  int grp  = lane >> 4, rl = lane & 15;

  const _Float16* Ab = A  + (size_t)m0 * K;
  const _Float16* Bb = BT + (size_t)n0 * K;
  int nch = K / BK;

  v8f c[4][2] = {};

#ifdef USE_TDM
  if (wid == 0) {
    tdm_load_tile(Ab, lds_off_u32(&ldsA[0][0][0]), K, BM);
    tdm_load_tile(Bb, lds_off_u32(&ldsB[0][0][0]), K, BN);
  }
  for (int ch = 0; ch < nch; ++ch) {
    int buf = ch & 1;
    if (wid == 0) {
      if (ch + 1 < nch) {
        tdm_load_tile(Ab + (size_t)(ch + 1) * BK, lds_off_u32(&ldsA[buf ^ 1][0][0]), K, BM);
        tdm_load_tile(Bb + (size_t)(ch + 1) * BK, lds_off_u32(&ldsB[buf ^ 1][0][0]), K, BN);
        __builtin_amdgcn_s_wait_tensorcnt((short)2);   // chunk ch landed
      } else {
        __builtin_amdgcn_s_wait_tensorcnt((short)0);
      }
    }
    __syncthreads();                                    // data ready for all waves
    v16h bfr0 = load_b_frag(&ldsB[buf][wn * 32 + rl][0], 0, grp);
    v16h bfr1 = load_b_frag(&ldsB[buf][wn * 32 + 16 + rl][0], 0, grp);
#pragma unroll
    for (int mi = 0; mi < 4; ++mi) {
      v16h a = load_a_frag(&ldsA[buf][wm * 64 + mi * 16 + rl][0], 0, grp);
      c[mi][0] = wmma_f16(a, bfr0, c[mi][0]);
      c[mi][1] = wmma_f16(a, bfr1, c[mi][1]);
    }
    __syncthreads();                                    // buffer reusable
  }
#else
  for (int ch = 0; ch < nch; ++ch) {
    int buf = ch & 1;
    int k0 = ch * BK;
    // cooperative staging: 128 rows x 32 halves per tile, 16B per thread-iter
    for (int i = tid; i < (BM * BK) / 8; i += 256) {
      int row = i >> 2, seg = i & 3;
      *(v8h*)&ldsA[buf][row][seg * 8] = *(const v8h*)(Ab + (size_t)row * K + k0 + seg * 8);
      *(v8h*)&ldsB[buf][row][seg * 8] = *(const v8h*)(Bb + (size_t)row * K + k0 + seg * 8);
    }
    __syncthreads();
    v16h bfr0 = load_b_frag(&ldsB[buf][wn * 32 + rl][0], 0, grp);
    v16h bfr1 = load_b_frag(&ldsB[buf][wn * 32 + 16 + rl][0], 0, grp);
#pragma unroll
    for (int mi = 0; mi < 4; ++mi) {
      v16h a = load_a_frag(&ldsA[buf][wm * 64 + mi * 16 + rl][0], 0, grp);
      c[mi][0] = wmma_f16(a, bfr0, c[mi][0]);
      c[mi][1] = wmma_f16(a, bfr1, c[mi][1]);
    }
    __syncthreads();
  }
#endif

  for (int mi = 0; mi < 4; ++mi)
    for (int ni = 0; ni < 2; ++ni)
      for (int r = 0; r < 8; ++r) {
        int m = m0 + wm * 64 + mi * 16 + grp * 8 + r;
        int n = n0 + wn * 32 + ni * 16 + rl;
        float val = c[mi][ni][r] + (bias ? bias[n] : 0.f);
        if (epi <= 1) {                       // q / k : [b,h,s,dh]
          int b_ = m >> 8, s_ = m & 255, h_ = n >> 6, dh = n & 63;
          out16[((size_t)(b_ * NHEAD + h_) * SEQ + s_) * DHEAD + dh] = (_Float16)val;
        } else if (epi == 2) {                // vT : [b,h,dh,s]
          int b_ = m >> 8, s_ = m & 255, h_ = n >> 6, dh = n & 63;
          out16[((size_t)(b_ * NHEAD + h_) * DHEAD + dh) * SEQ + s_] = (_Float16)val;
        } else if (epi == 3) {                // residual accumulate
          outF[(size_t)m * N + n] += val;
        } else {                              // exact GELU -> f16
          float ge = 0.5f * val * (1.0f + erff(val * 0.70710678118654752f));
          out16[(size_t)m * N + n] = (_Float16)ge;
        }
      }
}

// ---------------------------------------------------------------------------
// Attention scores: per (b,h) 16x16 tile of (s_q, s_k), K = DH = 64 via WMMA.
// Fuses 1/sqrt(DH) scale + causal mask. scores fp32 [b,h,s_q,s_k]
// ---------------------------------------------------------------------------
__global__ __launch_bounds__(256) void attn_score_kernel(
    const _Float16* __restrict__ q, const _Float16* __restrict__ k,
    float* __restrict__ scores)
{
  int lane = threadIdx.x & 31;
  int wave = blockIdx.x * 8 + (threadIdx.x >> 5);
  const int tilesPer = (SEQ / 16) * (SEQ / 16);       // 256
  int bh = wave / tilesPer;
  int t  = wave - bh * tilesPer;
  int tq = t / (SEQ / 16), tk = t - tq * (SEQ / 16);
  int sq0 = tq << 4, sk0 = tk << 4;
  int grp = lane >> 4, rl = lane & 15;

  const _Float16* qb = q + (size_t)bh * SEQ * DHEAD + (size_t)(sq0 + rl) * DHEAD;
  const _Float16* kb = k + (size_t)bh * SEQ * DHEAD + (size_t)(sk0 + rl) * DHEAD;

  v8f c = {};
  for (int k0 = 0; k0 < DHEAD; k0 += 32) {
    v16h a = load_a_frag(qb, k0, grp);
    v16h b = load_b_frag(kb, k0, grp);
    c = wmma_f16(a, b, c);
  }
  const float scale = 0.125f;                         // 1/sqrt(64)
  float* sb = scores + (size_t)bh * SEQ * SEQ;
  for (int r = 0; r < 8; ++r) {
    int sq = sq0 + grp * 8 + r;
    int sk = sk0 + rl;
    sb[(size_t)sq * SEQ + sk] = c[r] * scale + (sq >= sk ? 0.f : -1e10f);
  }
}

// ---------------------------------------------------------------------------
// Row softmax (256 cols), one wave32 per row, output f16 probs
// ---------------------------------------------------------------------------
__global__ __launch_bounds__(256) void softmax_kernel(
    const float* __restrict__ scores, _Float16* __restrict__ probs)
{
  int lane = threadIdx.x & 31;
  int row  = blockIdx.x * 8 + (threadIdx.x >> 5);
  const float* sr = scores + (size_t)row * SEQ;
  float v[8], mx = -3.0e38f;
  for (int i = 0; i < 8; ++i) { v[i] = sr[lane + 32 * i]; mx = fmaxf(mx, v[i]); }
  for (int off = 16; off > 0; off >>= 1) mx = fmaxf(mx, __shfl_xor(mx, off, 32));
  float sum = 0.f;
  for (int i = 0; i < 8; ++i) { v[i] = __expf(v[i] - mx); sum += v[i]; }
  for (int off = 16; off > 0; off >>= 1) sum += __shfl_xor(sum, off, 32);
  float inv = 1.0f / sum;
  _Float16* pr = probs + (size_t)row * SEQ;
  for (int i = 0; i < 8; ++i) pr[lane + 32 * i] = (_Float16)(v[i] * inv);
}

// ---------------------------------------------------------------------------
// o = probs @ v : per (b,h) 16x16 tile of (s_q, dh), K = S = 256 via WMMA.
// vT is [b,h,dh,s] so B-fragment reads are contiguous. Output f16 [m, D].
// ---------------------------------------------------------------------------
__global__ __launch_bounds__(256) void attn_av_kernel(
    const _Float16* __restrict__ probs, const _Float16* __restrict__ vT,
    _Float16* __restrict__ o)
{
  int lane = threadIdx.x & 31;
  int wave = blockIdx.x * 8 + (threadIdx.x >> 5);
  const int tilesPer = (SEQ / 16) * (DHEAD / 16);     // 64
  int bh = wave / tilesPer;
  int t  = wave - bh * tilesPer;
  int tq = t / (DHEAD / 16), td = t - tq * (DHEAD / 16);
  int sq0 = tq << 4, d0 = td << 4;
  int grp = lane >> 4, rl = lane & 15;

  const _Float16* pb = probs + (size_t)bh * SEQ * SEQ + (size_t)(sq0 + rl) * SEQ;
  const _Float16* vb = vT + (size_t)bh * DHEAD * SEQ + (size_t)(d0 + rl) * SEQ;

  v8f c = {};
  for (int k0 = 0; k0 < SEQ; k0 += 32) {
    v16h a = load_a_frag(pb, k0, grp);
    v16h b = load_b_frag(vb, k0, grp);
    c = wmma_f16(a, b, c);
  }
  int b_ = bh / NHEAD, h_ = bh - b_ * NHEAD;
  for (int r = 0; r < 8; ++r) {
    int sq = sq0 + grp * 8 + r;
    int dh = d0 + rl;
    o[((size_t)(b_ * SEQ + sq)) * DMODEL + h_ * DHEAD + dh] = (_Float16)c[r];
  }
}

// ---------------------------------------------------------------------------
// Final: gather last valid token, LayerNorm, project D->E (tiny, scalar)
// ---------------------------------------------------------------------------
__global__ __launch_bounds__(256) void final_kernel(
    const float* __restrict__ h, const int* __restrict__ text_len,
    const float* __restrict__ g, const float* __restrict__ bta,
    const float* __restrict__ Wout, float* __restrict__ out)
{
  __shared__ float xs[DMODEL];
  __shared__ float red[256];
  int b = blockIdx.x, tid = threadIdx.x;
  int last = text_len[b] - 1;
  if (last < 0) last = 0;
  if (last >= SEQ) last = SEQ - 1;
  const float* xr = h + ((size_t)b * SEQ + last) * DMODEL;
  float s = 0.f;
  for (int d = tid; d < DMODEL; d += 256) s += xr[d];
  red[tid] = s; __syncthreads();
  for (int off = 128; off > 0; off >>= 1) {
    if (tid < off) red[tid] += red[tid + off];
    __syncthreads();
  }
  float mu = red[0] * (1.0f / DMODEL);
  __syncthreads();
  float v = 0.f;
  for (int d = tid; d < DMODEL; d += 256) { float t = xr[d] - mu; v += t * t; }
  red[tid] = v; __syncthreads();
  for (int off = 128; off > 0; off >>= 1) {
    if (tid < off) red[tid] += red[tid + off];
    __syncthreads();
  }
  float rs = rsqrtf(red[0] * (1.0f / DMODEL) + 1e-5f);
  for (int d = tid; d < DMODEL; d += 256)
    xs[d] = (xr[d] - mu) * rs * g[d] + bta[d];
  __syncthreads();
  for (int n = tid; n < EMB; n += 256) {
    float acc = 0.f;
    for (int d = 0; d < DMODEL; ++d) acc += xs[d] * Wout[(size_t)d * EMB + n];
    out[(size_t)b * EMB + n] = acc;
  }
}

// ---------------------------------------------------------------------------
extern "C" void kernel_launch(void* const* d_in, const int* in_sizes, int n_in,
                              void* d_out, int out_size, void* d_ws, size_t ws_size,
                              hipStream_t stream) {
  const int*   text     = (const int*)  d_in[0];
  const int*   text_len = (const int*)  d_in[1];
  const float* w_voc    = (const float*)d_in[2];
  const float* w_pos    = (const float*)d_in[3];
  const float* ln1_g    = (const float*)d_in[4];
  const float* ln1_b    = (const float*)d_in[5];
  const float* Wq       = (const float*)d_in[6];
  const float* bq       = (const float*)d_in[7];
  const float* Wk       = (const float*)d_in[8];
  const float* Wv       = (const float*)d_in[9];
  const float* bv       = (const float*)d_in[10];
  const float* Wc       = (const float*)d_in[11];
  const float* bc       = (const float*)d_in[12];
  const float* ln2_g    = (const float*)d_in[13];
  const float* ln2_b    = (const float*)d_in[14];
  const float* W1       = (const float*)d_in[15];
  const float* b1       = (const float*)d_in[16];
  const float* W2       = (const float*)d_in[17];
  const float* b2       = (const float*)d_in[18];
  const float* lnf_g    = (const float*)d_in[19];
  const float* lnf_b    = (const float*)d_in[20];
  const float* Wout     = (const float*)d_in[21];
  float* out = (float*)d_out;

  char* ws = (char*)d_ws;
  float*    h       = (float*)ws;    ws += (size_t)MTOK * DMODEL * 4;
  _Float16* r16     = (_Float16*)ws; ws += (size_t)MTOK * DMODEL * 2;
  _Float16* act16   = (_Float16*)ws; ws += (size_t)MTOK * DFF * 2;
  _Float16* q16     = (_Float16*)ws; ws += (size_t)MTOK * DMODEL * 2;
  _Float16* k16     = (_Float16*)ws; ws += (size_t)MTOK * DMODEL * 2;
  _Float16* vT16    = (_Float16*)ws; ws += (size_t)MTOK * DMODEL * 2;
  float*    scores  = (float*)ws;    ws += (size_t)BATCH * NHEAD * SEQ * SEQ * 4;
  _Float16* probs16 = (_Float16*)ws; ws += (size_t)BATCH * NHEAD * SEQ * SEQ * 2;
  _Float16* wT16    = (_Float16*)ws; ws += (size_t)DFF * DMODEL * 2;

  const int convDD   = (DMODEL * DMODEL) / 256;
  const int convDF   = (DMODEL * DFF) / 256;
  const int gemmDD   = (MTOK / BM) * (DMODEL / BN);        // 16*8  = 128 blocks
  const int gemmDF   = (MTOK / BM) * (DFF / BN);           // 16*32 = 512 blocks
  const int scoreBlk = (BATCH * NHEAD * (SEQ / 16) * (SEQ / 16)) / 8;   // 4096
  const int smaxBlk  = (BATCH * NHEAD * SEQ) / 8;                       // 4096
  const int avBlk    = (BATCH * NHEAD * (SEQ / 16) * (DHEAD / 16)) / 8; // 1024

  embed_kernel<<<(MTOK * DMODEL + 255) / 256, 256, 0, stream>>>(text, w_voc, w_pos, h);

  for (int l = 0; l < NLAYER; ++l) {
    const float* wq = Wq + (size_t)l * DMODEL * DMODEL;
    const float* wk = Wk + (size_t)l * DMODEL * DMODEL;
    const float* wv = Wv + (size_t)l * DMODEL * DMODEL;
    const float* wc = Wc + (size_t)l * DMODEL * DMODEL;
    const float* w1 = W1 + (size_t)l * DMODEL * DFF;
    const float* w2 = W2 + (size_t)l * DFF * DMODEL;

    // --- attention resblock ---
    layernorm_f16_kernel<<<MTOK, 256, 0, stream>>>(h, ln1_g + l * DMODEL, ln1_b + l * DMODEL, r16);

    convert_wT_kernel<<<convDD, 256, 0, stream>>>(wq, wT16, DMODEL, DMODEL);
    gemm_f16_wmma_kernel<<<gemmDD, 256, 0, stream>>>(r16, wT16, bq + l * DMODEL, nullptr, q16, DMODEL, DMODEL, 0);

    convert_wT_kernel<<<convDD, 256, 0, stream>>>(wk, wT16, DMODEL, DMODEL);
    gemm_f16_wmma_kernel<<<gemmDD, 256, 0, stream>>>(r16, wT16, nullptr, nullptr, k16, DMODEL, DMODEL, 1);

    convert_wT_kernel<<<convDD, 256, 0, stream>>>(wv, wT16, DMODEL, DMODEL);
    gemm_f16_wmma_kernel<<<gemmDD, 256, 0, stream>>>(r16, wT16, bv + l * DMODEL, nullptr, vT16, DMODEL, DMODEL, 2);

    attn_score_kernel<<<scoreBlk, 256, 0, stream>>>(q16, k16, scores);
    softmax_kernel<<<smaxBlk, 256, 0, stream>>>(scores, probs16);
    attn_av_kernel<<<avBlk, 256, 0, stream>>>(probs16, vT16, act16);

    convert_wT_kernel<<<convDD, 256, 0, stream>>>(wc, wT16, DMODEL, DMODEL);
    gemm_f16_wmma_kernel<<<gemmDD, 256, 0, stream>>>(act16, wT16, bc + l * DMODEL, h, nullptr, DMODEL, DMODEL, 3);

    // --- MLP resblock ---
    layernorm_f16_kernel<<<MTOK, 256, 0, stream>>>(h, ln2_g + l * DMODEL, ln2_b + l * DMODEL, r16);

    convert_wT_kernel<<<convDF, 256, 0, stream>>>(w1, wT16, DMODEL, DFF);
    gemm_f16_wmma_kernel<<<gemmDF, 256, 0, stream>>>(r16, wT16, b1 + l * DFF, nullptr, act16, DFF, DMODEL, 4);

    convert_wT_kernel<<<convDF, 256, 0, stream>>>(w2, wT16, DFF, DMODEL);
    gemm_f16_wmma_kernel<<<gemmDD, 256, 0, stream>>>(act16, wT16, b2 + l * DMODEL, h, nullptr, DMODEL, DFF, 3);
  }

  final_kernel<<<BATCH, 256, 0, stream>>>(h, text_len, lnf_g, lnf_b, Wout, out);
}
